// GCN_45921790329163
// MI455X (gfx1250) — matrix operand order repaired and checked by reading, not compiled, not run on hardware
//
#include <hip/hip_runtime.h>
#include <cstdint>
#include <cstddef>

// ---------------------------------------------------------------------------
// Types / helpers
// ---------------------------------------------------------------------------
typedef __attribute__((ext_vector_type(16))) _Float16 v16h;
typedef __attribute__((ext_vector_type(8)))  _Float16 v8h;
typedef __attribute__((ext_vector_type(8)))  float    v8f;

#define DEV __device__ __forceinline__

#define Nn   4096
#define Tt   512
#define Fin  16
#define Hh   64
#define HIDc 256
#define NCLS 10

DEV v8f wmma_f16(v8f c, v16h a, v16h b) {
  // D = A(16x32 f16) * B(32x16 f16) + C(16x16 f32)
  return __builtin_amdgcn_wmma_f32_16x16x32_f16(false, a, false, b,
                                                (short)0, c, false, false);
}

// A operand fragment (16x32, f16, row-major source with leading dim ldh).
// Per ISA: lanes 0-15 hold rows M=0..15 with K {0..7, 16..23};
//          lanes 16-31 hold same rows with K {8..15, 24..31}.
DEV v16h load_a_frag(const _Float16* base, int ldh, int lane) {
  const int row = lane & 15;
  const int klo = (lane >> 4) * 8;
  const _Float16* p = base + row * ldh;
  v8h lo = *(const v8h*)(p + klo);
  v8h hi = *(const v8h*)(p + 16 + klo);
  v16h r;
#pragma unroll
  for (int i = 0; i < 8; ++i) { r[i] = lo[i]; r[8 + i] = hi[i]; }
  return r;
}

// B operand fragment (32x16). Source stored TRANSPOSED (Bt: N rows x K cols,
// row-major, leading dim ldh): lane n(=lane&15) holds column n,
// K {kb..kb+7, 16+kb..16+kb+7} with kb = (lane>>4)*8 (same as A).
DEV v16h load_b_frag(const _Float16* base, int ldh, int lane) {
  const int col = lane & 15;
  const int kb  = (lane >> 4) * 8;
  const _Float16* p = base + col * ldh + kb;
  v8h lo = *(const v8h*)(p);
  v8h hi = *(const v8h*)(p + 16);
  v16h r;
#pragma unroll
  for (int i = 0; i < 8; ++i) { r[i] = lo[i]; r[8 + i] = hi[i]; }
  return r;
}

DEV float sigm(float x)      { return 1.0f / (1.0f + __expf(-x)); }
DEV float tanh_fast(float x) { return 2.0f / (1.0f + __expf(-2.0f * x)) - 1.0f; }

// ---------------------------------------------------------------------------
// Kernel: features (N, F, T) f32  ->  (T, N, F) f16   (coalesced LSTM loads)
// ---------------------------------------------------------------------------
__global__ __launch_bounds__(256) void k_feat(const float* __restrict__ f,
                                              _Float16* __restrict__ o) {
  size_t idx = (size_t)blockIdx.x * 256 + threadIdx.x;   // over N*F*T
  if (idx >= (size_t)Nn * Fin * Tt) return;
  int t = (int)(idx % Tt);
  size_t nf = idx / Tt;
  int fc = (int)(nf % Fin);
  int n  = (int)(nf / Fin);
  o[((size_t)t * Nn + n) * Fin + fc] = (_Float16)f[idx];
}

// ---------------------------------------------------------------------------
// Kernel: 2-layer LSTM, persistent per-block over all 512 steps.
// Block = 4 waves (128 thr) owning NR=16 batch rows; wave w owns hidden tile
// ht=w*16 and computes all four 16x16 gate tiles, so the cell update is pure
// register math. ALL weight B-fragments (28 x v16h) live in VGPRs (>256 VGPRs
// per wave via gfx1250 VGPR-MSB); the serial loop's LDS traffic is only the
// recurrent h fragments. 4-wave blocks always fit at 1 wave/SIMD.
// ---------------------------------------------------------------------------
#define NR 16
__global__ __launch_bounds__(128) void k_lstm(
    const _Float16* __restrict__ featTN,               // (T, N, Fin) f16
    const float* __restrict__ wih0, const float* __restrict__ whh0,
    const float* __restrict__ bih0, const float* __restrict__ bhh0,
    const float* __restrict__ wih1, const float* __restrict__ whh1,
    const float* __restrict__ bih1, const float* __restrict__ bhh1,
    float* __restrict__ xseq)                          // (N, T) f32, h1 ch0
{
  __shared__ __align__(16) _Float16 sW[256 * 64];      // staging, reused 4x
  __shared__ float sB0[256], sB1[256];
  __shared__ __align__(16) _Float16 sH0[NR * 64], sH1[NR * 64];
  __shared__ __align__(16) _Float16 sXt[NR * 32];

  const int tid = threadIdx.x;
  const int lane = tid & 31, wv = tid >> 5;
  const int ht = wv * 16;                 // hidden tile (0,16,32,48)
  const int n0 = blockIdx.x * NR;
  const int mloc = (lane >> 4) * 8;       // accumulator slot v -> row mloc+v
  const int nloc = ht + (lane & 15);      // hidden index

  if (tid < 256) { sB0[tid] = bih0[tid] + bhh0[tid]; sB1[tid] = bih1[tid] + bhh1[tid]; }
  for (int i = tid; i < NR * 64; i += 128) { sH0[i] = (_Float16)0.f; sH1[i] = (_Float16)0.f; }
  for (int i = tid; i < NR * 32; i += 128) sXt[i] = (_Float16)0.f;   // K-pad stays 0

  // ---- hoist all weight B-fragments into registers (constant over t) ----
  v16h wIh0[4], wHh0[4][2], wIh1[4][2], wHh1[4][2];
  // Wih0 (4H x Fin), pad K 16 -> 32
  for (int i = tid; i < 256 * 32; i += 128) {
    int n = i >> 5, k = i & 31;
    sW[i] = (_Float16)(k < Fin ? wih0[n * Fin + k] : 0.0f);
  }
  __syncthreads();
#pragma unroll
  for (int g = 0; g < 4; ++g) wIh0[g] = load_b_frag(sW + (g * 64 + ht) * 32, 32, lane);
  __syncthreads();
  // Whh0 (4H x H)
  for (int i = tid; i < 256 * 64; i += 128) sW[i] = (_Float16)whh0[i];
  __syncthreads();
#pragma unroll
  for (int g = 0; g < 4; ++g) {
    wHh0[g][0] = load_b_frag(sW + (g * 64 + ht) * 64, 64, lane);
    wHh0[g][1] = load_b_frag(sW + (g * 64 + ht) * 64 + 32, 64, lane);
  }
  __syncthreads();
  // Wih1 (4H x H)
  for (int i = tid; i < 256 * 64; i += 128) sW[i] = (_Float16)wih1[i];
  __syncthreads();
#pragma unroll
  for (int g = 0; g < 4; ++g) {
    wIh1[g][0] = load_b_frag(sW + (g * 64 + ht) * 64, 64, lane);
    wIh1[g][1] = load_b_frag(sW + (g * 64 + ht) * 64 + 32, 64, lane);
  }
  __syncthreads();
  // Whh1 (4H x H)
  for (int i = tid; i < 256 * 64; i += 128) sW[i] = (_Float16)whh1[i];
  __syncthreads();
#pragma unroll
  for (int g = 0; g < 4; ++g) {
    wHh1[g][0] = load_b_frag(sW + (g * 64 + ht) * 64, 64, lane);
    wHh1[g][1] = load_b_frag(sW + (g * 64 + ht) * 64 + 32, 64, lane);
  }
  __syncthreads();

  float bias0[4], bias1[4];
#pragma unroll
  for (int g = 0; g < 4; ++g) { bias0[g] = sB0[g * 64 + nloc]; bias1[g] = sB1[g * 64 + nloc]; }

  v8f c0 = {}; v8f c1 = {};               // cell states (register-resident)

  for (int t = 0; t < Tt; ++t) {
    // stage x_t tile (16 rows x 16 features; cols 16..31 stay zero)
    const _Float16* ft = featTN + ((size_t)t * Nn + n0) * Fin;
    for (int i = tid; i < NR * Fin; i += 128) {
      int r = i >> 4, fc = i & 15;
      sXt[r * 32 + fc] = ft[r * Fin + fc];
    }
    if (tid < 8 && t + 1 < Tt)            // prefetch next step's tile (512B)
      __builtin_prefetch((const char*)(ft + (size_t)Nn * Fin) + tid * 64, 0, 1);
    __syncthreads();                      // (e) Xt visible

    // ---- layer 0: gates = Xt @ Wih0^T + H0 @ Whh0^T ----
    v16h aX   = load_a_frag(sXt, 32, lane);
    v16h aH0a = load_a_frag(sH0, 64, lane);
    v16h aH0b = load_a_frag(sH0 + 32, 64, lane);
    v8f acc[4];
#pragma unroll
    for (int g = 0; g < 4; ++g) {
      v8f a = {};
      a = wmma_f16(a, aX,   wIh0[g]);
      a = wmma_f16(a, aH0a, wHh0[g][0]);
      a = wmma_f16(a, aH0b, wHh0[g][1]);
      acc[g] = a;
    }
    __syncthreads();                      // (a) all reads of old H0 done
#pragma unroll
    for (int v = 0; v < 8; ++v) {
      float i_ = sigm(acc[0][v] + bias0[0]);
      float f_ = sigm(acc[1][v] + bias0[1]);
      float g_ = tanh_fast(acc[2][v] + bias0[2]);
      float o_ = sigm(acc[3][v] + bias0[3]);
      float c  = f_ * c0[v] + i_ * g_;
      c0[v] = c;
      sH0[(mloc + v) * 64 + nloc] = (_Float16)(o_ * tanh_fast(c));
    }
    __syncthreads();                      // (b) new H0 visible

    // ---- layer 1: gates = H0 @ Wih1^T + H1 @ Whh1^T ----
    v16h aXa  = load_a_frag(sH0, 64, lane);
    v16h aXb  = load_a_frag(sH0 + 32, 64, lane);
    v16h aH1a = load_a_frag(sH1, 64, lane);
    v16h aH1b = load_a_frag(sH1 + 32, 64, lane);
#pragma unroll
    for (int g = 0; g < 4; ++g) {
      v8f a = {};
      a = wmma_f16(a, aXa,  wIh1[g][0]);
      a = wmma_f16(a, aXb,  wIh1[g][1]);
      a = wmma_f16(a, aH1a, wHh1[g][0]);
      a = wmma_f16(a, aH1b, wHh1[g][1]);
      acc[g] = a;
    }
    __syncthreads();                      // (c) all reads of old H1 done
#pragma unroll
    for (int v = 0; v < 8; ++v) {
      float i_ = sigm(acc[0][v] + bias1[0]);
      float f_ = sigm(acc[1][v] + bias1[1]);
      float g_ = tanh_fast(acc[2][v] + bias1[2]);
      float o_ = sigm(acc[3][v] + bias1[3]);
      float c  = f_ * c1[v] + i_ * g_;
      c1[v] = c;
      float h = o_ * tanh_fast(c);
      sH1[(mloc + v) * 64 + nloc] = (_Float16)h;
      if (nloc == 0) xseq[(size_t)(n0 + mloc + v) * Tt + t] = h;  // channel 0
    }
    // no trailing barrier needed: next-step staging only writes sXt, whose
    // readers all passed barrier (c); H1 readers pass next step's (b).
  }
}

// ---------------------------------------------------------------------------
// Kernel: per-row mean/std of xseq (N,T); emit f16 copies (raw + centered)
// ---------------------------------------------------------------------------
__global__ __launch_bounds__(256) void k_rowstats(const float* __restrict__ xseq,
                                                  float* __restrict__ stdv,
                                                  _Float16* __restrict__ xh,
                                                  _Float16* __restrict__ xch) {
  const int n = blockIdx.x, tid = threadIdx.x;
  __shared__ float r1[256], r2[256];
  const float* row = xseq + (size_t)n * Tt;
  float s = 0.f, s2 = 0.f;
  for (int t = tid; t < Tt; t += 256) { float v = row[t]; s += v; s2 += v * v; }
  r1[tid] = s; r2[tid] = s2; __syncthreads();
  for (int off = 128; off; off >>= 1) {
    if (tid < off) { r1[tid] += r1[tid + off]; r2[tid] += r2[tid + off]; }
    __syncthreads();
  }
  const float mu  = r1[0] * (1.0f / Tt);
  const float var = r2[0] * (1.0f / Tt) - mu * mu;
  if (tid == 0) stdv[n] = sqrtf(fmaxf(var, 0.f));
  for (int t = tid; t < Tt; t += 256) {
    float v = row[t];
    xh [(size_t)n * Tt + t] = (_Float16)v;
    xch[(size_t)n * Tt + t] = (_Float16)(v - mu);
  }
}

// ---------------------------------------------------------------------------
// Kernel: rowsum of adj (f16) -> d = rs>0 ? rs^-1/2 : 0
// ---------------------------------------------------------------------------
__global__ __launch_bounds__(256) void k_rowsum(const _Float16* __restrict__ adj,
                                                float* __restrict__ dvec) {
  const int n = blockIdx.x, tid = threadIdx.x;
  __shared__ float red[256];
  const _Float16* row = adj + (size_t)n * Nn;
  float s = 0.f;
  for (int j = tid; j < Nn; j += 256) s += (float)row[j];
  red[tid] = s; __syncthreads();
  for (int off = 128; off; off >>= 1) {
    if (tid < off) red[tid] += red[tid + off];
    __syncthreads();
  }
  if (tid == 0) { float rs = red[0]; dvec[n] = rs > 0.f ? rsqrtf(rs) : 0.f; }
}

// ---------------------------------------------------------------------------
// Kernel: transpose f32 (R x C) -> f16 (C x R), optional per-src-row scale
// ---------------------------------------------------------------------------
__global__ __launch_bounds__(256) void k_tran(const float* __restrict__ src,
                                              _Float16* __restrict__ dst,
                                              int R, int C,
                                              const float* __restrict__ scale) {
  size_t idx = (size_t)blockIdx.x * 256 + threadIdx.x;
  if (idx >= (size_t)R * C) return;
  int r = (int)(idx / C), c = (int)(idx % C);
  float s = scale ? scale[r] : 1.0f;
  dst[(size_t)c * R + r] = (_Float16)(s * src[idx]);
}

// ---------------------------------------------------------------------------
// Generic WMMA GEMM:  C(MxN) = A(MxK, f16 row-major) * Bt^T
//   (Bt stored N x K row-major, f16)
// Block tile 128x64, BK=32, 8 waves: wave w -> 16-row strip, 4 col tiles.
// DOUBLE-BUFFERED async staging: both A and B tiles are fetched with
// global_load_async_to_lds_b128 (ASYNCcnt) into the alternate LDS buffer
// while WMMAs consume the current one; one s_wait_asynccnt + one barrier
// per K-step.
// EPI 0: store f32 (Cf)
// EPI 1: adjacency: adj = guard(cov/T / (si*sj)) + I  -> f16 Ch
// EPI 2: relu(d[m]*acc + bias[n])                     -> f16 Ch
// ---------------------------------------------------------------------------
template <int EPI>
__global__ __launch_bounds__(256) void k_gemm(
    const _Float16* __restrict__ A, int lda,
    const _Float16* __restrict__ Bt, int ldb, int K,
    float* __restrict__ Cf, _Float16* __restrict__ Ch, int ldc,
    const float* __restrict__ rowscale, const float* __restrict__ bias,
    float alpha) {
  constexpr int BM = 128, BN = 64, LDSD = 40;   // pad 32->40 halves: no conflicts
  __shared__ __align__(16) _Float16 sA[2][BM * LDSD];
  __shared__ __align__(16) _Float16 sB[2][BN * LDSD];
  const int tid = threadIdx.x, lane = tid & 31, wv = tid >> 5;
  const int m0 = blockIdx.y * BM, n0 = blockIdx.x * BN;
  v8f acc[4] = {};

  // issue ASYNCcnt-tracked global->LDS loads for one K-tile into buffer `buf`
  auto issue = [&](int buf, int k0) {
    // A tile: 128 rows x 4 chunks of 8 halves (2 per thread)
#pragma unroll
    for (int it = 0; it < 2; ++it) {
      const int i = tid + it * 256;
      const int r = i >> 2, c8 = (i & 3) * 8;
      uint32_t lptr = (uint32_t)(uintptr_t)(&sA[buf][r * LDSD + c8]);
      const _Float16* gptr = &A[(size_t)(m0 + r) * lda + k0 + c8];
      asm volatile("global_load_async_to_lds_b128 %0, %1, off"
                   :: "v"(lptr), "v"(gptr) : "memory");
    }
    // B tile: 64 rows x 4 chunks (1 per thread)
    {
      const int r = tid >> 2, c8 = (tid & 3) * 8;
      uint32_t lptr = (uint32_t)(uintptr_t)(&sB[buf][r * LDSD + c8]);
      const _Float16* gptr = &Bt[(size_t)(n0 + r) * ldb + k0 + c8];
      asm volatile("global_load_async_to_lds_b128 %0, %1, off"
                   :: "v"(lptr), "v"(gptr) : "memory");
    }
  };

  const int nk = K >> 5;
  issue(0, 0);                                           // prologue
  for (int kk = 0; kk < nk; ++kk) {
    const int cur = kk & 1;
    asm volatile("s_wait_asynccnt 0x0" ::: "memory");    // my chunks landed
    __syncthreads();                                     // everyone's landed
    if (kk + 1 < nk) issue(cur ^ 1, (kk + 1) * 32);      // overlap next tile

    v16h af = load_a_frag(sA[cur] + (wv * 16) * LDSD, LDSD, lane);
#pragma unroll
    for (int ct = 0; ct < 4; ++ct)
      acc[ct] = wmma_f16(acc[ct], af,
                         load_b_frag(sB[cur] + (ct * 16) * LDSD, LDSD, lane));
    // no trailing barrier: next iteration's wait+barrier orders reuse of the
    // buffer written by `issue` (disjoint from `cur`) against this compute.
  }

  const int mbase = m0 + wv * 16 + (lane >> 4) * 8;
  const int nlo   = lane & 15;
#pragma unroll
  for (int ct = 0; ct < 4; ++ct) {
    const int n = n0 + ct * 16 + nlo;
#pragma unroll
    for (int v = 0; v < 8; ++v) {
      const int m = mbase + v;
      float val = acc[ct][v];
      if (EPI == 0) {
        Cf[(size_t)m * ldc + n] = val;
      } else if (EPI == 1) {
        float si = rowscale[m], sj = rowscale[n];
        float c = val * alpha;                        // cov = (xc@xc^T)/T
        float corr = (si == 0.f || sj == 0.f) ? 0.f : c / (si * sj);
        if (m == n) corr += 1.f;                      // + eye
        Ch[(size_t)m * ldc + n] = (_Float16)corr;
      } else {
        float r = rowscale[m] * val + bias[n];        // D*(adj@(D*y)) + b
        Ch[(size_t)m * ldc + n] = (_Float16)(r > 0.f ? r : 0.f);
      }
    }
  }
}

// ---------------------------------------------------------------------------
// Kernel: classifier  out(N x 10) = agg2(f16) @ clf_w(f32) + clf_b
// ---------------------------------------------------------------------------
__global__ __launch_bounds__(256) void k_clf(const _Float16* __restrict__ agg,
                                             const float* __restrict__ w,
                                             const float* __restrict__ b,
                                             float* __restrict__ out) {
  const int n = blockIdx.x, tid = threadIdx.x;
  __shared__ float row[HIDc];
  row[tid] = (float)agg[(size_t)n * HIDc + tid];
  __syncthreads();
  if (tid < NCLS) {
    float s = b[tid];
    for (int k = 0; k < HIDc; ++k) s += row[k] * w[k * NCLS + tid];
    out[n * NCLS + tid] = s;
  }
}

// ---------------------------------------------------------------------------
// Workspace layout (bytes). [0,64MB) staged features, reused after LSTM.
// ---------------------------------------------------------------------------
static constexpr size_t OFF_FEATH = 0;             // (T,N,F) f16 : 64 MB
static constexpr size_t OFF_ADJ   = 0;             // adj f16     : 32 MB (reuse)
static constexpr size_t OFF_XH    = 33554432;      // x f16 (N,T) :  4 MB
static constexpr size_t OFF_XCH   = 37748736;      // xc f16      :  4 MB
static constexpr size_t OFF_Y     = 41943040;      // y f32 (N,256): 4 MB
static constexpr size_t OFF_DYT   = 46137344;      // dy^T f16    :  2 MB
static constexpr size_t OFF_A1    = 48234496;      // agg1 f16    :  2 MB
static constexpr size_t OFF_A2    = 50331648;      // agg2 f16    :  2 MB
static constexpr size_t OFF_XSEQ  = 67108864;      // xseq f32    :  8 MB
static constexpr size_t OFF_STD   = 75497472;      // std f32
static constexpr size_t OFF_D     = 75513856;      // d   f32
static constexpr size_t OFF_G1T   = 75530240;      // gc1_w^T f16 : 256 KB
static constexpr size_t OFF_G2T   = 75792384;      // gc2_w^T f16 : 128 KB

extern "C" void kernel_launch(void* const* d_in, const int* in_sizes, int n_in,
                              void* d_out, int out_size, void* d_ws, size_t ws_size,
                              hipStream_t stream) {
  (void)in_sizes; (void)n_in; (void)out_size; (void)ws_size;
  const float* features = (const float*)d_in[0];
  const float* wih0 = (const float*)d_in[1];
  const float* whh0 = (const float*)d_in[2];
  const float* bih0 = (const float*)d_in[3];
  const float* bhh0 = (const float*)d_in[4];
  const float* wih1 = (const float*)d_in[5];
  const float* whh1 = (const float*)d_in[6];
  const float* bih1 = (const float*)d_in[7];
  const float* bhh1 = (const float*)d_in[8];
  const float* gc1w = (const float*)d_in[9];
  const float* gc1b = (const float*)d_in[10];
  const float* gc2w = (const float*)d_in[11];
  const float* gc2b = (const float*)d_in[12];
  const float* clfw = (const float*)d_in[13];
  const float* clfb = (const float*)d_in[14];
  float* out = (float*)d_out;

  char* ws = (char*)d_ws;
  _Float16* featH = (_Float16*)(ws + OFF_FEATH);
  _Float16* adj   = (_Float16*)(ws + OFF_ADJ);
  _Float16* xh    = (_Float16*)(ws + OFF_XH);
  _Float16* xch   = (_Float16*)(ws + OFF_XCH);
  float*    ybuf  = (float*)   (ws + OFF_Y);
  _Float16* dyT   = (_Float16*)(ws + OFF_DYT);
  _Float16* a1    = (_Float16*)(ws + OFF_A1);
  _Float16* a2    = (_Float16*)(ws + OFF_A2);
  float*    xseq  = (float*)   (ws + OFF_XSEQ);
  float*    stdv  = (float*)   (ws + OFF_STD);
  float*    dvec  = (float*)   (ws + OFF_D);
  _Float16* g1t   = (_Float16*)(ws + OFF_G1T);
  _Float16* g2t   = (_Float16*)(ws + OFF_G2T);

  // 1) stage features (N,F,T)f32 -> (T,N,F)f16
  k_feat<<<(Nn * Fin * Tt) / 256, 256, 0, stream>>>(features, featH);

  // 2) persistent 2-layer LSTM (256 blocks x 4 waves, 16 batch rows each)
  k_lstm<<<Nn / NR, 128, 0, stream>>>(featH, wih0, whh0, bih0, bhh0,
                                      wih1, whh1, bih1, bhh1, xseq);

  // 3) per-row stats + f16 copies of x / centered x
  k_rowstats<<<Nn, 256, 0, stream>>>(xseq, stdv, xh, xch);

  // 4) adjacency: adj = corr(xc@xc^T / T) + I   (xc is both A and Bt: symmetric)
  k_gemm<1><<<dim3(Nn / 64, Nn / 128), 256, 0, stream>>>(
      xch, Tt, xch, Tt, Tt, nullptr, adj, Nn, stdv, nullptr, 1.0f / Tt);

  // 5) degree vector d = rowsum(adj)^-1/2 (guarded)
  k_rowsum<<<Nn, 256, 0, stream>>>(adj, dvec);

  // 6) GCN weight transposes to f16 (N x K convention for the B operand)
  k_tran<<<(Tt * HIDc) / 256, 256, 0, stream>>>(gc1w, g1t, Tt, HIDc, nullptr);
  k_tran<<<(HIDc * HIDc) / 256, 256, 0, stream>>>(gc2w, g2t, HIDc, HIDc, nullptr);

  // 7) GCN layer 1: y1 = x @ gc1_w ; agg1 = relu(D*(adj@(D*y1)) + b1)
  k_gemm<0><<<dim3(HIDc / 64, Nn / 128), 256, 0, stream>>>(
      xh, Tt, g1t, Tt, Tt, ybuf, nullptr, HIDc, nullptr, nullptr, 0.f);
  k_tran<<<(Nn * HIDc) / 256, 256, 0, stream>>>(ybuf, dyT, Nn, HIDc, dvec);
  k_gemm<2><<<dim3(HIDc / 64, Nn / 128), 256, 0, stream>>>(
      adj, Nn, dyT, Nn, Nn, nullptr, a1, HIDc, dvec, gc1b, 0.f);

  // 8) GCN layer 2
  k_gemm<0><<<dim3(HIDc / 64, Nn / 128), 256, 0, stream>>>(
      a1, HIDc, g2t, HIDc, HIDc, ybuf, nullptr, HIDc, nullptr, nullptr, 0.f);
  k_tran<<<(Nn * HIDc) / 256, 256, 0, stream>>>(ybuf, dyT, Nn, HIDc, dvec);
  k_gemm<2><<<dim3(HIDc / 64, Nn / 128), 256, 0, stream>>>(
      adj, Nn, dyT, Nn, Nn, nullptr, a2, HIDc, dvec, gc2b, 0.f);

  // 9) classifier
  k_clf<<<Nn, 256, 0, stream>>>(a2, clfw, clfb, out);
}
